// CausalMultiHeadSelfAttention_24816321036590
// MI455X (gfx1250) — matrix operand order repaired and checked by reading, not compiled, not run on hardware
//
#include <hip/hip_runtime.h>

// ---------------------------------------------------------------------------
// CausalMultiHeadSelfAttention  B=4 S=2048 D=1024 H=16 DK=64  (MI455X gfx1250)
// bf16 WMMA everywhere, flash-attention fused core, fp32 accumulate.
// TDM (tensor_load_to_lds) stages the K tile in the attention kernel.
// ---------------------------------------------------------------------------

typedef __attribute__((ext_vector_type(16))) __bf16 v16bf;
typedef __attribute__((ext_vector_type(8)))  float  v8f;
typedef __attribute__((ext_vector_type(4)))  unsigned int v4u;
typedef __attribute__((ext_vector_type(8)))  int    v8i;
typedef __attribute__((ext_vector_type(4)))  int    v4i;

#define S_LEN  2048
#define D_DIM  1024
#define H_NUM  16
#define DK_DIM 64
#define M_ROWS 8192   // B*S

#if defined(__has_builtin)
#  if __has_builtin(__builtin_amdgcn_tensor_load_to_lds)
#    define USE_TDM 1
#  else
#    define USE_TDM 0
#  endif
#else
#  define USE_TDM 0
#endif

union Frag16 { v16bf v; float4 q[2]; };
union Frag8  { v8f   v; float  f[8]; };

__device__ __forceinline__ unsigned short f32_to_bf16_bits(float f) {
  unsigned int u   = __float_as_uint(f);
  unsigned int lsb = (u >> 16) & 1u;
  return (unsigned short)((u + 0x7FFFu + lsb) >> 16);   // round-to-nearest-even
}

__device__ __forceinline__ v8f zero8() {
  v8f z = {0.f, 0.f, 0.f, 0.f, 0.f, 0.f, 0.f, 0.f};
  return z;
}

// Build a 16-element bf16 fragment from two 16-byte chunks.
__device__ __forceinline__ v16bf frag_from2(const unsigned short* lo,
                                            const unsigned short* hi) {
  Frag16 u;
  u.q[0] = *(const float4*)lo;
  u.q[1] = *(const float4*)hi;
  return u.v;
}

// ---------------------------------------------------------------------------
// fp32 -> bf16 conversion
// ---------------------------------------------------------------------------
__global__ __launch_bounds__(256) void cvt_f32_to_bf16(
    const float* __restrict__ src, unsigned short* __restrict__ dst, int n) {
  int i = (blockIdx.x * 256 + threadIdx.x) * 4;
  if (i < n) {
    float4 v = *(const float4*)(src + i);
    dst[i + 0] = f32_to_bf16_bits(v.x);
    dst[i + 1] = f32_to_bf16_bits(v.y);
    dst[i + 2] = f32_to_bf16_bits(v.z);
    dst[i + 3] = f32_to_bf16_bits(v.w);
  }
}

// ---------------------------------------------------------------------------
// NT GEMM:  Y[M,N] = A[M,K] * W[N,K]^T   (bf16 in, f32 accumulate)
// MODE 0: store bf16 into [B,H,S,DK] head layout (QKV projections)
// MODE 1: store f32 row-major (final output projection)
// Wave tile 32x64 (2x4 WMMA), block tile 64x256 (8 waves as 2x4).
// ---------------------------------------------------------------------------
template <int MODE>
__global__ __launch_bounds__(256) void gemm_nt_bf16(
    const unsigned short* __restrict__ A,
    const unsigned short* __restrict__ W,
    void* __restrict__ Y, int M, int N, int K) {
  const int tid   = threadIdx.x;
  const int lane  = tid & 31;
  const int w     = tid >> 5;
  const int l16   = lane & 15;
  const int half  = lane >> 4;
  const int mbase = blockIdx.y * 64  + (w & 1)  * 32;
  const int nbase = blockIdx.x * 256 + (w >> 1) * 64;

  Frag8 acc[2][4];
#pragma unroll
  for (int i = 0; i < 2; ++i)
#pragma unroll
    for (int j = 0; j < 4; ++j) acc[i][j].v = zero8();

#pragma unroll 2
  for (int k = 0; k < K; k += 32) {
    if (k + 128 < K) {
      __builtin_prefetch(A + (size_t)(mbase + l16) * K + k + 128, 0, 0);
      __builtin_prefetch(W + (size_t)(nbase + l16) * K + k + 128, 0, 0);
    }
    v16bf af[2], bf[4];
#pragma unroll
    for (int i = 0; i < 2; ++i) {
      // A-frag 16x32: lane row = l16, 16 contiguous K at half*16
      const unsigned short* p = A + (size_t)(mbase + i * 16 + l16) * K + k + half * 16;
      af[i] = frag_from2(p, p + 8);
    }
#pragma unroll
    for (int j = 0; j < 4; ++j) {
      // B-frag 32x16: lane col = l16, 16 contiguous K at half*16
      const unsigned short* p = W + (size_t)(nbase + j * 16 + l16) * K + k + half * 16;
      bf[j] = frag_from2(p, p + 8);
    }
#pragma unroll
    for (int i = 0; i < 2; ++i)
#pragma unroll
      for (int j = 0; j < 4; ++j)
        acc[i][j].v = __builtin_amdgcn_wmma_f32_16x16x32_bf16(
            false, af[i], false, bf[j], (short)0, acc[i][j].v, false, false);
  }

  // Epilogue. C/D layout: VGPR r -> row = half*8 + r, lane l16 -> col.
#pragma unroll
  for (int i = 0; i < 2; ++i) {
#pragma unroll
    for (int j = 0; j < 4; ++j) {
      const int col = nbase + j * 16 + l16;
#pragma unroll
      for (int r = 0; r < 8; ++r) {
        const int row = mbase + i * 16 + half * 8 + r;
        const float val = acc[i][j].f[r];
        if (MODE == 0) {
          const int bb = row >> 11, ss = row & 2047;
          const int hh = col >> 6,  dk = col & 63;
          ((unsigned short*)Y)[(((size_t)(bb * H_NUM + hh) * S_LEN) + ss) * DK_DIM + dk] =
              f32_to_bf16_bits(val);
        } else {
          ((float*)Y)[(size_t)row * N + col] = val;
        }
      }
    }
  }
}

// ---------------------------------------------------------------------------
// Fused causal flash attention.
// grid = (S/128, B*H), block = 256 (8 waves). Wave owns 16 query rows.
// K tile (32x64 bf16, row-major) staged by the Tensor Data Mover; V tile
// staged cooperatively with an in-flight transpose for contiguous B-frags.
// ---------------------------------------------------------------------------
__global__ __launch_bounds__(256) void attn_fa(
    const unsigned short* __restrict__ Qh,
    const unsigned short* __restrict__ Kh,
    const unsigned short* __restrict__ Vh,
    unsigned short* __restrict__ attnB) {
  __shared__ __align__(16) unsigned short Kt[32 * 64];       // [key][dk]
  __shared__ __align__(16) unsigned short Vt[64 * 32];       // [dk][key] (transposed)
  __shared__ __align__(16) unsigned short Pw[8 * 16 * 32];   // per-wave P buffer

  const int tid  = threadIdx.x;
  const int lane = tid & 31;
  const int w    = tid >> 5;
  const int l16  = lane & 15;
  const int half = lane >> 4;
  const int bh   = blockIdx.y;                 // b*H + h
  const int b    = bh >> 4;
  const int h    = bh & 15;
  const int qbase = blockIdx.x * 128 + w * 16; // wave's first query row in S

  const unsigned short* Qg = Qh + (size_t)bh * S_LEN * DK_DIM;
  const unsigned short* Kg = Kh + (size_t)bh * S_LEN * DK_DIM;
  const unsigned short* Vg = Vh + (size_t)bh * S_LEN * DK_DIM;

  // Q A-frags (16 x 64 split into two 16x32 chunks), resident for whole loop.
  v16bf aq[2];
#pragma unroll
  for (int c = 0; c < 2; ++c) {
    const unsigned short* p = Qg + (size_t)(qbase + l16) * DK_DIM + c * 32 + half * 16;
    aq[c] = frag_from2(p, p + 8);
  }

  Frag8 o[4];
#pragma unroll
  for (int d = 0; d < 4; ++d) o[d].v = zero8();
  float mrow[8], lrow[8];
#pragma unroll
  for (int r = 0; r < 8; ++r) { mrow[r] = -__builtin_inff(); lrow[r] = 0.f; }

  const int kb_last = blockIdx.x * 4 + 3;  // inclusive: cover queries of whole block
  const int qmax    = qbase + 15;

  for (int kb = 0; kb <= kb_last; ++kb) {
    // ---- stage K tile via TDM (wave 0 issues the DMA) ----
#if USE_TDM
    if (w == 0) {
      const unsigned long long ga =
          (unsigned long long)(uintptr_t)(Kg + (size_t)(kb * 32) * DK_DIM);
      const unsigned lds_off = (unsigned)(uintptr_t)(&Kt[0]);
      v4u g0;
      g0[0] = 1u;                                       // count=1, user descriptor
      g0[1] = lds_off;                                  // lds_addr (bytes)
      g0[2] = (unsigned)(ga & 0xffffffffu);             // global_addr[31:0]
      g0[3] = (unsigned)((ga >> 32) & 0x1ffffffu)       // global_addr[56:32]
              | (2u << 30);                             // type=2 ("image")
      v8i g1;
      g1[0] = (int)(1u << 16);       // workgroup_mask=0, data_size=1 (2 bytes)
      g1[1] = (int)(64u << 16);      // tensor_dim0[15:0]=64  (bits 63:48)
      g1[2] = (int)(2048u << 16);    // tensor_dim0 hi=0, tensor_dim1[15:0]=2048
      g1[3] = (int)(64u << 16);      // tensor_dim1 hi=0, tile_dim0=64
      g1[4] = 32;                    // tile_dim1=32, tile_dim2=0
      g1[5] = 64;                    // tensor_dim0_stride=64 (low 32)
      g1[6] = 0;                     // stride hi, tensor_dim1_stride lo
      g1[7] = 0;
      v4i gz = {0, 0, 0, 0};
#if __clang_major__ >= 23
      v8i gz8 = {0, 0, 0, 0, 0, 0, 0, 0};
      __builtin_amdgcn_tensor_load_to_lds(g0, g1, gz, gz, gz8, 0);
#else
      __builtin_amdgcn_tensor_load_to_lds(g0, g1, gz, gz, 0);
#endif
#if __has_builtin(__builtin_amdgcn_s_wait_tensorcnt)
      __builtin_amdgcn_s_wait_tensorcnt(0);
#else
      asm volatile("s_wait_tensorcnt 0x0" ::: "memory");
#endif
    }
#endif
    // ---- cooperative stage of V (transposed) into LDS ----
    {
      const int idx = tid * 8;            // 256 threads * 8 bf16 = 2048 elems
      const int row = idx >> 6;           // key within block
      const int col = idx & 63;           // dk
#if !USE_TDM
      const float4 kv = *(const float4*)(Kg + (size_t)(kb * 32 + row) * DK_DIM + col);
      *(float4*)(&Kt[row * 64 + col]) = kv;
#endif
      const float4 vv = *(const float4*)(Vg + (size_t)(kb * 32 + row) * DK_DIM + col);
      const unsigned short* ve = (const unsigned short*)&vv;
#pragma unroll
      for (int i = 0; i < 8; ++i) Vt[(col + i) * 32 + row] = ve[i];
    }
    __syncthreads();

    if (kb * 32 <= qmax) {  // wave-uniform: EXEC stays all-ones for WMMA
      // ---- scores = Q * K^T : two 16x16 tiles, each 2 WMMAs over dk ----
      Frag8 sc[2];
      sc[0].v = zero8(); sc[1].v = zero8();
#pragma unroll
      for (int t = 0; t < 2; ++t)
#pragma unroll
        for (int c = 0; c < 2; ++c) {
          const unsigned short* p = &Kt[(t * 16 + l16) * 64 + c * 32 + half * 16];
          v16bf bk = frag_from2(p, p + 8);
          sc[t].v = __builtin_amdgcn_wmma_f32_16x16x32_bf16(
              false, aq[c], false, bk, (short)0, sc[t].v, false, false);
        }

      // ---- scale by 1/sqrt(64) and apply causal mask ----
#pragma unroll
      for (int t = 0; t < 2; ++t)
#pragma unroll
        for (int r = 0; r < 8; ++r) {
          const int kabs = kb * 32 + t * 16 + l16;
          const int qabs = qbase + half * 8 + r;
          const float sv = sc[t].f[r] * 0.125f;
          sc[t].f[r] = (kabs <= qabs) ? sv : -__builtin_inff();
        }

      // ---- online softmax: row max / sum across the 16 lanes of a row ----
      float pm[8];
#pragma unroll
      for (int r = 0; r < 8; ++r) pm[r] = fmaxf(sc[0].f[r], sc[1].f[r]);
#pragma unroll
      for (int off = 8; off >= 1; off >>= 1)
#pragma unroll
        for (int r = 0; r < 8; ++r)
          pm[r] = fmaxf(pm[r], __shfl_xor(pm[r], off, 16));

      Frag8 p0, p1;
      float psum[8], lsc[8];
#pragma unroll
      for (int r = 0; r < 8; ++r) {
        const float mnew = fmaxf(mrow[r], pm[r]);
        lsc[r]  = __expf(mrow[r] - mnew);
        p0.f[r] = __expf(sc[0].f[r] - mnew);
        p1.f[r] = __expf(sc[1].f[r] - mnew);
        psum[r] = p0.f[r] + p1.f[r];
        mrow[r] = mnew;
      }
#pragma unroll
      for (int off = 8; off >= 1; off >>= 1)
#pragma unroll
        for (int r = 0; r < 8; ++r)
          psum[r] += __shfl_xor(psum[r], off, 16);
#pragma unroll
      for (int r = 0; r < 8; ++r) lrow[r] = lrow[r] * lsc[r] + psum[r];
#pragma unroll
      for (int d = 0; d < 4; ++d)
#pragma unroll
        for (int r = 0; r < 8; ++r) o[d].f[r] *= lsc[r];

      // ---- re-swizzle P (C-layout -> A-layout) via wave-private LDS ----
      unsigned short* pp = &Pw[w * 16 * 32];
#pragma unroll
      for (int r = 0; r < 8; ++r) {
        pp[(half * 8 + r) * 32 + l16]      = f32_to_bf16_bits(p0.f[r]);
        pp[(half * 8 + r) * 32 + 16 + l16] = f32_to_bf16_bits(p1.f[r]);
      }
      asm volatile("s_wait_dscnt 0x0" ::: "memory");
      v16bf pa = frag_from2(&pp[l16 * 32 + half * 16], &pp[l16 * 32 + half * 16 + 8]);

      // ---- O += P * V : four 16x16 dk tiles ----
#pragma unroll
      for (int d = 0; d < 4; ++d) {
        const unsigned short* p = &Vt[(d * 16 + l16) * 32 + half * 16];
        v16bf bv = frag_from2(p, p + 8);
        o[d].v = __builtin_amdgcn_wmma_f32_16x16x32_bf16(
            false, pa, false, bv, (short)0, o[d].v, false, false);
      }
    }
    __syncthreads();
  }

  // ---- normalize and write merged-head bf16 output [B,S,D] ----
  float inv[8];
#pragma unroll
  for (int r = 0; r < 8; ++r) inv[r] = 1.0f / lrow[r];
#pragma unroll
  for (int d = 0; d < 4; ++d)
#pragma unroll
    for (int r = 0; r < 8; ++r) {
      const int s   = qbase + half * 8 + r;
      const int col = h * 64 + d * 16 + l16;
      attnB[((size_t)(b * S_LEN + s)) * D_DIM + col] =
          f32_to_bf16_bits(o[d].f[r] * inv[r]);
    }
}

// ---------------------------------------------------------------------------
// Launch
// ---------------------------------------------------------------------------
extern "C" void kernel_launch(void* const* d_in, const int* in_sizes, int n_in,
                              void* d_out, int out_size, void* d_ws, size_t ws_size,
                              hipStream_t stream) {
  (void)in_sizes; (void)n_in; (void)out_size; (void)ws_size;
  const float* x  = (const float*)d_in[0];
  const float* Wq = (const float*)d_in[1];
  const float* Wk = (const float*)d_in[2];
  const float* Wv = (const float*)d_in[3];
  const float* Wo = (const float*)d_in[4];

  const size_t NX = (size_t)M_ROWS * D_DIM;   // 8,388,608
  const size_t NW = (size_t)D_DIM * D_DIM;    // 1,048,576

  unsigned short* xb  = (unsigned short*)d_ws;
  unsigned short* wqb = xb  + NX;
  unsigned short* wkb = wqb + NW;
  unsigned short* wvb = wkb + NW;
  unsigned short* wob = wvb + NW;
  unsigned short* qh  = wob + NW;   // [B,H,S,DK] bf16
  unsigned short* kh  = qh  + NX;
  unsigned short* vh  = kh  + NX;
  unsigned short* ab  = vh  + NX;   // attention output, [B,S,D] bf16

  cvt_f32_to_bf16<<<(int)(NX / 1024), 256, 0, stream>>>(x,  xb,  (int)NX);
  cvt_f32_to_bf16<<<(int)(NW / 1024), 256, 0, stream>>>(Wq, wqb, (int)NW);
  cvt_f32_to_bf16<<<(int)(NW / 1024), 256, 0, stream>>>(Wk, wkb, (int)NW);
  cvt_f32_to_bf16<<<(int)(NW / 1024), 256, 0, stream>>>(Wv, wvb, (int)NW);
  cvt_f32_to_bf16<<<(int)(NW / 1024), 256, 0, stream>>>(Wo, wob, (int)NW);

  dim3 gg(D_DIM / 256, M_ROWS / 64);  // (4, 128)
  gemm_nt_bf16<0><<<gg, 256, 0, stream>>>(xb, wqb, qh, M_ROWS, D_DIM, D_DIM);
  gemm_nt_bf16<0><<<gg, 256, 0, stream>>>(xb, wkb, kh, M_ROWS, D_DIM, D_DIM);
  gemm_nt_bf16<0><<<gg, 256, 0, stream>>>(xb, wvb, vh, M_ROWS, D_DIM, D_DIM);

  attn_fa<<<dim3(S_LEN / 128, 4 * H_NUM), 256, 0, stream>>>(qh, kh, vh, ab);

  gemm_nt_bf16<1><<<gg, 256, 0, stream>>>(ab, wob, d_out, M_ROWS, D_DIM, D_DIM);
}
